// PagedJointAttention_46926812676617
// MI455X (gfx1250) — matrix operand order repaired and checked by reading, not compiled, not run on hardware
//
#include <hip/hip_runtime.h>

typedef unsigned int u32;
typedef __attribute__((ext_vector_type(16))) __bf16 v16bf;
typedef __attribute__((ext_vector_type(8)))  float  v8f;
typedef __attribute__((ext_vector_type(4)))  u32    v4u;
typedef __attribute__((ext_vector_type(8)))  int    v8i;
typedef __attribute__((ext_vector_type(4)))  int    v4i;

#define E_DIM 2048
#define HEADS 16
#define HD    128
#define LQ    1280   // S + S_enc
#define LKV   3328   // S_cache + S + S_enc
#define S_DEC 1024
#define S_ENC 256
#define BATCH 2

// ---------------- bf16 helpers ----------------
__device__ __forceinline__ unsigned short f32_bf16(float f) {
  u32 u = __builtin_bit_cast(u32, f);
  u += 0x7FFFu + ((u >> 16) & 1u);            // round-to-nearest-even
  return (unsigned short)(u >> 16);
}
__device__ __forceinline__ float bf16_f32(unsigned short h) {
  return __builtin_bit_cast(float, (u32)h << 16);
}

union FragU { v16bf v; uint4 q[2]; unsigned short s[16]; };

__device__ __forceinline__ v8f wmma_bf16(v16bf a, v16bf b, v8f c) {
  return __builtin_amdgcn_wmma_f32_16x16x32_bf16(false, a, false, b, (short)0, c,
                                                 false, false);
}

// A fragment (16 rows x 32 K) from a row-major [row][k] buffer, stride in halves.
__device__ __forceinline__ v16bf load_a_frag(const unsigned short* base, int stride, int lane) {
  int r  = lane & 15;
  int c0 = (lane & 16) ? 16 : 0;
  const uint4* p = (const uint4*)(base + r * stride + c0);
  FragU u; u.q[0] = p[0]; u.q[1] = p[1];      // +8 halves
  return u.v;
}
// B fragment (32 K x 16 N) from an [n][k] (k-contiguous) buffer, stride in halves.
__device__ __forceinline__ v16bf load_b_frag_kcontig(const unsigned short* base, int stride, int lane) {
  int n  = lane & 15;
  int k0 = (lane & 16) ? 16 : 0;
  const uint4* p = (const uint4*)(base + n * stride + k0);
  FragU u; u.q[0] = p[0]; u.q[1] = p[1];
  return u.v;
}
// B fragment from a [k][n] row-major buffer (e.g. V tile), row stride in halves.
__device__ __forceinline__ v16bf load_b_frag_nmajor(const unsigned short* base, int stride, int lane) {
  int n  = lane & 15;
  int k0 = (lane & 16) ? 16 : 0;
  FragU u;
#pragma unroll
  for (int i = 0; i < 16; ++i) u.s[i] = base[(k0 + i) * stride + n];
  return u.v;
}

// ------------- TDM: load a [rows x 128 bf16] tile from global into LDS -------------
// D# per ISA ch.8: group0 {count|lds_addr|global_addr|type=2}, group1 {data_size=2B,
// tensor_dim0=128, tile_dim0=128, tile_dim1=rows, dim0_stride=128}. Groups 2/3 unused (2-D).
__device__ __forceinline__ void tdm_load_tile(const unsigned short* gsrc, u32 lds_byte_off,
                                              int rows, int rows_left) {
  size_t ga = (size_t)gsrc;
  v4u g0;
  g0[0] = 1u;                                   // count=1, is_restore=0, gather off
  g0[1] = lds_byte_off;                         // lds_addr
  g0[2] = (u32)ga;                              // global_addr[31:0]
  g0[3] = ((u32)(ga >> 32) & 0x01FFFFFFu) | (2u << 30);  // addr[56:32] | type=2
  v8i g1;
  g1[0] = (int)(1u << 16);                      // data_size=1 (2 bytes)
  g1[1] = (int)(128u << 16);                    // tensor_dim0 = 128 (bits 79:48)
  g1[2] = (int)((u32)rows_left << 16);          // tensor_dim1 (bits 111:80)
  g1[3] = (int)(128u << 16);                    // tile_dim0 = 128 (bits 127:112)
  g1[4] = rows;                                 // tile_dim1 = rows, tile_dim2 = 0
  g1[5] = 128;                                  // tensor_dim0_stride = 128
  g1[6] = 0;
  g1[7] = 0;
  v4i z4 = {0, 0, 0, 0};
  v8i z8 = {0, 0, 0, 0, 0, 0, 0, 0};
  __builtin_amdgcn_tensor_load_to_lds(g0, g1, z4, z4, z8, 0);   // 6-arg (clang-23) form
}

// ---------------- conversion kernels ----------------
__global__ void cvt_f32_bf16_kernel(const float* __restrict__ s,
                                    unsigned short* __restrict__ d, int n) {
  int i = blockIdx.x * 256 + threadIdx.x;
  if (i < n) d[i] = f32_bf16(s[i]);
}

// W [k][n] f32 -> Wt [n][k] bf16, 64x64 tiles through LDS
__global__ __launch_bounds__(256)
void transpose_cvt_kernel(const float* __restrict__ src, unsigned short* __restrict__ dst) {
  __shared__ unsigned short tile[64][65];
  const int t = threadIdx.x;
  const int bn = blockIdx.x * 64;   // n
  const int bk = blockIdx.y * 64;   // k
#pragma unroll
  for (int i = 0; i < 16; ++i) {
    int e = t + i * 256;
    int k = e >> 6, n = e & 63;     // coalesced on n
    tile[n][k] = f32_bf16(src[(size_t)(bk + k) * E_DIM + bn + n]);
  }
  __syncthreads();
#pragma unroll
  for (int i = 0; i < 16; ++i) {
    int e = t + i * 256;
    int n = e >> 6, k = e & 63;     // coalesced on k
    dst[(size_t)(bn + n) * E_DIM + bk + k] = tile[n][k];
  }
}

// k_cache/v_cache [B,H,2048,128] f32 -> rows 0..2047 of [B,H,3328,128] bf16
__global__ void cvt_cache_kernel(const float* __restrict__ s,
                                 unsigned short* __restrict__ d, int n) {
  int i = blockIdx.x * 256 + threadIdx.x;
  if (i >= n) return;
  int dd   = i & 127;
  int r    = i >> 7;
  int srow = r & 2047;
  int bh   = r >> 11;
  d[((size_t)bh * LKV + srow) * HD + dd] = f32_bf16(s[i]);
}

// ---------------- RMSNorm (one wave per 128-elem row) ----------------
__global__ __launch_bounds__(256)
void rmsnorm_kernel(unsigned short* __restrict__ buf,
                    const float* __restrict__ g1, const float* __restrict__ g2,
                    int L, int split, int nrows) {
  int row  = blockIdx.x * 8 + (threadIdx.x >> 5);
  int lane = threadIdx.x & 31;
  if (row >= nrows) return;
  unsigned short* p = buf + (size_t)row * HD;
  float v[4]; float ss = 0.f;
#pragma unroll
  for (int i = 0; i < 4; ++i) { v[i] = bf16_f32(p[lane * 4 + i]); ss += v[i] * v[i]; }
#pragma unroll
  for (int m = 16; m >= 1; m >>= 1) ss += __shfl_xor(ss, m, 32);
  float sc = rsqrtf(ss * (1.f / 128.f) + 1e-6f);
  const float* g = ((row % L) < split) ? g1 : g2;
#pragma unroll
  for (int i = 0; i < 4; ++i) p[lane * 4 + i] = f32_bf16(v[i] * sc * g[lane * 4 + i]);
}

// ---- shared GEMM tile staging: A[128x32] row-major, Wt[n][k] -> ldsB[64x32] -------
__device__ __forceinline__ void stage_tiles(const unsigned short* __restrict__ Arow0,
                                            size_t a_stride_u4,   // uint4 per A row
                                            const unsigned short* __restrict__ Wt,
                                            int tile_n, int k0, int t,
                                            unsigned short* ldsA, unsigned short* ldsB) {
  // A: thread t owns half of row (t>>1): 2 contiguous b128 loads + b128 LDS stores
  {
    int m = t >> 1, seg = (t & 1) * 2;
    const uint4* srcp = (const uint4*)(Arow0 + (size_t)m * (a_stride_u4 * 8) + k0) ;
    uint4* dstp = (uint4*)(ldsA + m * 32);
    dstp[seg]     = srcp[seg];
    dstp[seg + 1] = srcp[seg + 1];
  }
  // B: thread t owns one b128 of Wt row (t>>2)
  {
    int n = t >> 2, q = t & 3;
    ((uint4*)(ldsB + n * 32))[q] =
        ((const uint4*)(Wt + (size_t)(tile_n + n) * E_DIM + k0))[q];
  }
}

// ---------------- projection GEMM: bf16 A[M,E] x bf16 Wt[E,E](pre-T) + bias ----------
__global__ __launch_bounds__(256)
void gemm_proj_kernel(const unsigned short* __restrict__ A,
                      const unsigned short* __restrict__ Wt,
                      const float* __restrict__ bias,
                      unsigned short* __restrict__ dst,
                      int Lsrc, int Ldst, int dst_off) {
  __shared__ unsigned short ldsA[128 * 32];   // [m][k]
  __shared__ unsigned short ldsB[64 * 32];    // [n][k]
  const int t = threadIdx.x, lane = t & 31, w = t >> 5, hi = lane >> 4;
  const int tile_m = blockIdx.y * 128;
  const int tile_n = blockIdx.x * 64;
  v8f acc[4] = {};
  const unsigned short* Arow0 = A + (size_t)tile_m * E_DIM;

  for (int k0 = 0; k0 < E_DIM; k0 += 32) {
    __syncthreads();
    stage_tiles(Arow0, E_DIM / 8, Wt, tile_n, k0, t, ldsA, ldsB);
    if (k0 + 32 < E_DIM) {                     // -> global_prefetch_b8
      __builtin_prefetch(Arow0 + (size_t)(t >> 1) * E_DIM + k0 + 32, 0, 1);
      __builtin_prefetch(Wt + (size_t)(tile_n + (t >> 2)) * E_DIM + k0 + 32, 0, 1);
    }
    __syncthreads();
    v16bf a = load_a_frag(ldsA + (w * 16) * 32, 32, lane);
#pragma unroll
    for (int nt = 0; nt < 4; ++nt) {
      v16bf b = load_b_frag_kcontig(ldsB + (nt * 16) * 32, 32, lane);
      acc[nt] = wmma_bf16(a, b, acc[nt]);
    }
  }
  // epilogue: +bias, scatter into [B,H,Ldst,d] with concat offset
#pragma unroll
  for (int nt = 0; nt < 4; ++nt)
#pragma unroll
    for (int v = 0; v < 8; ++v) {
      int m = tile_m + w * 16 + v + 8 * hi;
      int n = tile_n + nt * 16 + (lane & 15);
      float val = acc[nt][v] + bias[n];
      int b = m / Lsrc, s = m % Lsrc;
      int h = n >> 7, dd = n & 127;
      dst[(((size_t)(b * HEADS + h) * Ldst) + dst_off + s) * HD + dd] = f32_bf16(val);
    }
}

// ---------------- output GEMM: bf16 attnO x bf16 Wt(pre-T) + bias -> f32 out ---------
__global__ __launch_bounds__(256)
void gemm_out_kernel(const unsigned short* __restrict__ A,
                     const unsigned short* __restrict__ Wt,
                     const float* __restrict__ bias,
                     float* __restrict__ out, int Ssrc, int src_off) {
  __shared__ unsigned short ldsA[128 * 32];
  __shared__ unsigned short ldsB[64 * 32];
  const int t = threadIdx.x, lane = t & 31, w = t >> 5, hi = lane >> 4;
  const int tile_m = blockIdx.y * 128;
  const int tile_n = blockIdx.x * 64;
  v8f acc[4] = {};
  // row-mapped A base for this thread's staging row (decoder/encoder row selection)
  int gm_mine = tile_m + (threadIdx.x >> 1);
  int arow = (gm_mine / Ssrc) * LQ + src_off + (gm_mine % Ssrc);
  const unsigned short* Amine = A + (size_t)arow * E_DIM;

  for (int k0 = 0; k0 < E_DIM; k0 += 32) {
    __syncthreads();
    {
      int seg = (t & 1) * 2;
      const uint4* srcp = (const uint4*)(Amine + k0);
      uint4* dstp = (uint4*)(ldsA + (t >> 1) * 32);
      dstp[seg] = srcp[seg]; dstp[seg + 1] = srcp[seg + 1];
    }
    {
      int n = t >> 2, q = t & 3;
      ((uint4*)(ldsB + n * 32))[q] =
          ((const uint4*)(Wt + (size_t)(tile_n + n) * E_DIM + k0))[q];
    }
    if (k0 + 32 < E_DIM) {
      __builtin_prefetch(Amine + k0 + 32, 0, 1);
      __builtin_prefetch(Wt + (size_t)(tile_n + (t >> 2)) * E_DIM + k0 + 32, 0, 1);
    }
    __syncthreads();
    v16bf a = load_a_frag(ldsA + (w * 16) * 32, 32, lane);
#pragma unroll
    for (int nt = 0; nt < 4; ++nt) {
      v16bf b = load_b_frag_kcontig(ldsB + (nt * 16) * 32, 32, lane);
      acc[nt] = wmma_bf16(a, b, acc[nt]);
    }
  }
#pragma unroll
  for (int nt = 0; nt < 4; ++nt)
#pragma unroll
    for (int v = 0; v < 8; ++v) {
      int m = tile_m + w * 16 + v + 8 * hi;
      int n = tile_n + nt * 16 + (lane & 15);
      out[(size_t)m * E_DIM + n] = acc[nt][v] + bias[n];
    }
}

// ---------------- flash attention with TDM-staged K/V tiles ----------------
__global__ __launch_bounds__(256)
void attn_kernel(const unsigned short* __restrict__ Q,
                 const unsigned short* __restrict__ K,
                 const unsigned short* __restrict__ V,
                 unsigned short* __restrict__ O) {
  __shared__ unsigned short ldsK[64 * HD];        // 16 KB
  __shared__ unsigned short ldsV[64 * HD];        // 16 KB
  __shared__ unsigned short ldsP[8 * 16 * 64];    // 16 KB, per-wave private P
  const int t = threadIdx.x, lane = t & 31, w = t >> 5, hi = lane >> 4;
  const int bh    = blockIdx.x;                   // b*H + h
  const int qrow0 = blockIdx.y * 128 + w * 16;
  const unsigned short* Qb = Q + (size_t)bh * LQ * HD;
  const unsigned short* Kb = K + (size_t)bh * LKV * HD;
  const unsigned short* Vb = V + (size_t)bh * LKV * HD;
  const float scale = 0.08838834764831845f;       // 1/sqrt(128)
  const u32 ldsK_off = (u32)(size_t)(void*)ldsK;  // LDS aperture: low 32 bits = LDS addr
  const u32 ldsV_off = (u32)(size_t)(void*)ldsV;

  v16bf qf[4];                                    // Q 16x128 as 4 A-frags, resident
#pragma unroll
  for (int dc = 0; dc < 4; ++dc)
    qf[dc] = load_a_frag(Qb + (size_t)qrow0 * HD + dc * 32, HD, lane);

  v8f o[8] = {};
  float mrow[8], lrow[8];
#pragma unroll
  for (int v = 0; v < 8; ++v) { mrow[v] = -3.0e38f; lrow[v] = 0.f; }

  for (int kv = 0; kv < LKV; kv += 64) {
    __syncthreads();                              // previous tile fully consumed
    if (w == 0) {                                 // one wave drives the Tensor Data Mover
      tdm_load_tile(Kb + (size_t)kv * HD, ldsK_off, 64, LKV - kv);
      tdm_load_tile(Vb + (size_t)kv * HD, ldsV_off, 64, LKV - kv);
      __builtin_amdgcn_s_wait_tensorcnt(0);
    }
    __syncthreads();                              // publish LDS tiles to all waves

    // S = Q . K^T : 4 n-tiles x 4 d-chunks
    v8f s[4] = {};
#pragma unroll
    for (int nt = 0; nt < 4; ++nt)
#pragma unroll
      for (int dc = 0; dc < 4; ++dc) {
        v16bf b = load_b_frag_kcontig(ldsK + (nt * 16) * HD + dc * 32, HD, lane);
        s[nt] = wmma_bf16(qf[dc], b, s[nt]);
      }

    // online softmax per row; C-layout rows live within 16-lane halves
    unsigned short* P = ldsP + w * (16 * 64);
#pragma unroll
    for (int v = 0; v < 8; ++v) {
      float mx = fmaxf(fmaxf(s[0][v], s[1][v]), fmaxf(s[2][v], s[3][v])) * scale;
#pragma unroll
      for (int m = 1; m <= 8; m <<= 1) mx = fmaxf(mx, __shfl_xor(mx, m, 32));
      float mnew = fmaxf(mrow[v], mx);
      float sum = 0.f;
#pragma unroll
      for (int nt = 0; nt < 4; ++nt) {
        float pv = __expf(s[nt][v] * scale - mnew);
        sum += pv;
        P[(v + 8 * hi) * 64 + nt * 16 + (lane & 15)] = f32_bf16(pv);
      }
#pragma unroll
      for (int m = 1; m <= 8; m <<= 1) sum += __shfl_xor(sum, m, 32);
      float alpha = __expf(mrow[v] - mnew);
      lrow[v] = lrow[v] * alpha + sum;
      mrow[v] = mnew;
#pragma unroll
      for (int dt = 0; dt < 8; ++dt) o[dt][v] *= alpha;
    }

    // O += P . V  (same-wave LDS ops are in-order -> no barrier for private P)
#pragma unroll
    for (int kc = 0; kc < 2; ++kc) {
      v16bf a = load_a_frag(P + kc * 32, 64, lane);
#pragma unroll
      for (int dt = 0; dt < 8; ++dt) {
        v16bf b = load_b_frag_nmajor(ldsV + (kc * 32) * HD + dt * 16, HD, lane);
        o[dt] = wmma_bf16(a, b, o[dt]);
      }
    }
  }

  // epilogue: /l, write head-interleaved [B, LQ, E]
  const int b = bh >> 4, h = bh & 15;
#pragma unroll
  for (int dt = 0; dt < 8; ++dt)
#pragma unroll
    for (int v = 0; v < 8; ++v) {
      int q = qrow0 + v + 8 * hi;
      float val = o[dt][v] / lrow[v];
      O[((size_t)(b * LQ + q)) * E_DIM + h * HD + dt * 16 + (lane & 15)] = f32_bf16(val);
    }
}

// ---------------- host launch ----------------
extern "C" void kernel_launch(void* const* d_in, const int* in_sizes, int n_in,
                              void* d_out, int out_size, void* d_ws, size_t ws_size,
                              hipStream_t stream) {
  (void)in_sizes; (void)n_in; (void)out_size; (void)ws_size;
  const float* x   = (const float*)d_in[0];
  const float* kc  = (const float*)d_in[1];
  const float* vc  = (const float*)d_in[2];
  const float* enc = (const float*)d_in[3];
  const float* Wf[8] = { (const float*)d_in[4],  (const float*)d_in[6],
                         (const float*)d_in[8],  (const float*)d_in[10],
                         (const float*)d_in[12], (const float*)d_in[14],
                         (const float*)d_in[16], (const float*)d_in[18] };
  const float* bq  = (const float*)d_in[5];
  const float* bk  = (const float*)d_in[7];
  const float* bv  = (const float*)d_in[9];
  const float* baq = (const float*)d_in[11];
  const float* bak = (const float*)d_in[13];
  const float* bav = (const float*)d_in[15];
  const float* bo  = (const float*)d_in[17];
  const float* bao = (const float*)d_in[19];
  const float* gq  = (const float*)d_in[20];
  const float* gk  = (const float*)d_in[21];
  const float* gaq = (const float*)d_in[22];
  const float* gak = (const float*)d_in[23];

  size_t off = 0;
  auto alloc_us = [&](size_t elems) {
    unsigned short* p = (unsigned short*)((char*)d_ws + off);
    off += (elems * 2 + 255) & ~(size_t)255;
    return p;
  };
  const size_t EE = (size_t)E_DIM * E_DIM;
  unsigned short* Wb[8];                    // pre-transposed bf16 weights [n][k]
  for (int i = 0; i < 8; ++i) Wb[i] = alloc_us(EE);
  unsigned short* xb = alloc_us((size_t)BATCH * S_DEC * E_DIM);
  unsigned short* eb = alloc_us((size_t)BATCH * S_ENC * E_DIM);
  unsigned short* Qb = alloc_us((size_t)BATCH * HEADS * LQ * HD);
  unsigned short* Kb = alloc_us((size_t)BATCH * HEADS * LKV * HD);
  unsigned short* Vb = alloc_us((size_t)BATCH * HEADS * LKV * HD);
  unsigned short* Ob = alloc_us((size_t)BATCH * LQ * E_DIM);

  auto cvt = [&](const float* s, unsigned short* d, size_t n) {
    cvt_f32_bf16_kernel<<<dim3((unsigned)((n + 255) / 256)), dim3(256), 0, stream>>>(s, d, (int)n);
  };
  cvt(x,   xb, (size_t)BATCH * S_DEC * E_DIM);
  cvt(enc, eb, (size_t)BATCH * S_ENC * E_DIM);
  for (int i = 0; i < 8; ++i)               // convert + transpose weights
    transpose_cvt_kernel<<<dim3(32, 32), dim3(256), 0, stream>>>(Wf[i], Wb[i]);

  const int ncache = BATCH * HEADS * 2048 * HD;
  cvt_cache_kernel<<<dim3(ncache / 256), dim3(256), 0, stream>>>(kc, Kb, ncache);
  cvt_cache_kernel<<<dim3(ncache / 256), dim3(256), 0, stream>>>(vc, Vb, ncache);

  dim3 blk(256);
  // decoder projections (M = 2048): Q, then K/V at concat offset 2048
  gemm_proj_kernel<<<dim3(32, 16), blk, 0, stream>>>(xb, Wb[0], bq, Qb, S_DEC, LQ, 0);
  gemm_proj_kernel<<<dim3(32, 16), blk, 0, stream>>>(xb, Wb[1], bk, Kb, S_DEC, LKV, 2048);
  gemm_proj_kernel<<<dim3(32, 16), blk, 0, stream>>>(xb, Wb[2], bv, Vb, S_DEC, LKV, 2048);
  // encoder projections (M = 512): appended at offsets 1024 / 3072
  gemm_proj_kernel<<<dim3(32, 4), blk, 0, stream>>>(eb, Wb[3], baq, Qb, S_ENC, LQ, S_DEC);
  gemm_proj_kernel<<<dim3(32, 4), blk, 0, stream>>>(eb, Wb[4], bak, Kb, S_ENC, LKV, 3072);
  gemm_proj_kernel<<<dim3(32, 4), blk, 0, stream>>>(eb, Wb[5], bav, Vb, S_ENC, LKV, 3072);

  // post-concat RMSNorm on Q (gq|gaq split at 1024) and K (gk|gak split at 3072)
  rmsnorm_kernel<<<dim3(BATCH * HEADS * LQ / 8), blk, 0, stream>>>(Qb, gq, gaq, LQ, S_DEC,
                                                                   BATCH * HEADS * LQ);
  rmsnorm_kernel<<<dim3(BATCH * HEADS * LKV / 8), blk, 0, stream>>>(Kb, gk, gak, LKV, 3072,
                                                                    BATCH * HEADS * LKV);

  // flash attention: 32 (b,h) x 10 q-tiles
  attn_kernel<<<dim3(BATCH * HEADS, LQ / 128), blk, 0, stream>>>(Qb, Kb, Vb, Ob);

  // output projections straight into d_out (hidden, then enc_out)
  float* outp = (float*)d_out;
  gemm_out_kernel<<<dim3(32, 16), blk, 0, stream>>>(Ob, Wb[6], bo, outp, S_DEC, 0);
  gemm_out_kernel<<<dim3(32, 4),  blk, 0, stream>>>(Ob, Wb[7], bao,
                                                    outp + (size_t)BATCH * S_DEC * E_DIM,
                                                    S_ENC, S_DEC);
}
